// Block_70317204570642
// MI455X (gfx1250) — matrix operand order repaired and checked by reading, not compiled, not run on hardware
//
#include <hip/hip_runtime.h>
#include <hip/hip_bf16.h>
#include <math.h>

// ---------------------------------------------------------------------------
// Types / fragments for gfx1250 WMMA (wave32)
// ---------------------------------------------------------------------------
typedef __attribute__((ext_vector_type(16))) __bf16 v16bf;
typedef __attribute__((ext_vector_type(8)))  float  v8f;
typedef unsigned int u32x4 __attribute__((ext_vector_type(4)));
typedef int          i32x4 __attribute__((ext_vector_type(4)));
typedef int          i32x8 __attribute__((ext_vector_type(8)));

union Frag {
  v16bf v;
  uint4 q[2];   // 2 x 16B chunks (8 bf16 each)
};

__device__ __forceinline__ __bf16 f2bf(float f) { return (__bf16)f; }

__device__ __forceinline__ v8f wmma_bf16(const Frag& a, const Frag& b, v8f c) {
  // D = A(16x32 bf16) * B(32x16 bf16) + C(16x16 f32)
  return __builtin_amdgcn_wmma_f32_16x16x32_bf16(
      /*neg_a=*/false, a.v, /*neg_b=*/false, b.v,
      /*c_mod=*/(short)0, c, /*reuse_a=*/false, /*reuse_b=*/false);
}

#if __has_builtin(__builtin_amdgcn_tensor_load_to_lds) && \
    __has_builtin(__builtin_amdgcn_s_wait_tensorcnt)
#define USE_TDM 1
#else
#define USE_TDM 0
#endif

#if USE_TDM
// Issue one TDM 2D tile load: tile_d0 (contiguous, elems) x tile_d1 rows of
// bf16, row stride = stride_elems, into LDS at byte offset lds_off.
// D# layout per CDNA5 ISA ch.8: group0 {flags, lds_addr, global_addr, type=2},
// group1 {data_size=2B, tensor dims, tile dims, dim0 stride}.
// This toolchain exposes the 6-arg builtin:
//   (u32x4 g0, i32x8 g1, i32x4 g2, i32x4 g3, i32x8 g4, i32 cpol)
__device__ __forceinline__ void tdm_load_2d(unsigned lds_off, const __bf16* gptr,
                                            unsigned tensor_d0, unsigned tensor_d1,
                                            unsigned tile_d0, unsigned tile_d1,
                                            unsigned stride_elems) {
  unsigned long long ga = (unsigned long long)(uintptr_t)gptr;
  u32x4 g0;
  g0[0] = 1u;                                   // count=1, user descriptor
  g0[1] = lds_off;                              // lds_addr
  g0[2] = (unsigned)ga;                         // global_addr[31:0]
  g0[3] = (unsigned)(ga >> 32) | (2u << 30);    // global_addr[56:32] | type=2
  i32x8 g1;
  g1[0] = (int)(1u << 16);                      // workgroup_mask=0, data_size=1 (2B)
  g1[1] = (int)((tensor_d0 & 0xFFFFu) << 16);   // tensor_dim0[15:0] @ bits 63:48
  g1[2] = (int)((tensor_d0 >> 16) | ((tensor_d1 & 0xFFFFu) << 16));
  g1[3] = (int)((tensor_d1 >> 16) | (tile_d0 << 16));   // tile_dim0 @ 127:112
  g1[4] = (int)(tile_d1 & 0xFFFFu);             // tile_dim1, tile_dim2=0
  g1[5] = (int)stride_elems;                    // tensor_dim0_stride[31:0]
  g1[6] = 0;                                    // stride[47:32], dim1_stride lo
  g1[7] = 0;
  i32x4 z4 = {0, 0, 0, 0};
  i32x8 z8 = {0, 0, 0, 0, 0, 0, 0, 0};
  __builtin_amdgcn_tensor_load_to_lds(g0, g1, z4, z4, z8, 0);
}
#endif

// ---------------------------------------------------------------------------
// Problem constants
// ---------------------------------------------------------------------------
#define BATCH 4
#define TSEQ  2048
#define CEMB  1024
#define NHEAD 16
#define HDIM  64
#define MTOT  (BATCH * TSEQ)   // 8192 rows

// ---------------------------------------------------------------------------
// fp32 -> bf16 conversion (row-major copy)
// ---------------------------------------------------------------------------
__global__ void f32_to_bf16_kernel(const float* __restrict__ in,
                                   __bf16* __restrict__ out, int n) {
  int i = blockIdx.x * blockDim.x + threadIdx.x;
  if (i < n) out[i] = f2bf(in[i]);
}

// fp32 [R, Ncol] -> bf16 [Ncol, R]  (transpose, so GEMM B operand is [N,K])
__global__ void f32_to_bf16_T_kernel(const float* __restrict__ in,
                                     __bf16* __restrict__ out, int R, int Ncol) {
  long i = (long)blockIdx.x * blockDim.x + threadIdx.x;
  long total = (long)R * Ncol;
  if (i >= total) return;
  int r = (int)(i / Ncol);
  int c = (int)(i % Ncol);
  out[(size_t)c * R + r] = f2bf(in[i]);
}

// bf16 V [B*T, H*64] -> bf16 Vt [B*H*64, T]
__global__ void transpose_v_kernel(const __bf16* __restrict__ V,
                                   __bf16* __restrict__ Vt) {
  long i = (long)blockIdx.x * blockDim.x + threadIdx.x;
  long total = (long)MTOT * CEMB;
  if (i >= total) return;
  int d  = (int)(i & 63);
  long r = i >> 6;               // (b*T+t)*H + h
  int h  = (int)(r & (NHEAD - 1));
  long bt = r >> 4;              // b*T + t
  int t  = (int)(bt & (TSEQ - 1));
  int b  = (int)(bt >> 11);
  Vt[(((size_t)b * NHEAD + h) * HDIM + d) * TSEQ + t] = V[i];
}

// ---------------------------------------------------------------------------
// LayerNorm (fp32 in) -> bf16 out.  One 256-thread block per row, C=1024.
// ---------------------------------------------------------------------------
__global__ void ln_bf16_kernel(const float* __restrict__ x,
                               const float* __restrict__ g,
                               const float* __restrict__ be,
                               __bf16* __restrict__ out) {
  const int C = CEMB;
  int row = blockIdx.x, tid = threadIdx.x;
  const float4 val = ((const float4*)(x + (size_t)row * C))[tid];
  float s  = val.x + val.y + val.z + val.w;
  float s2 = val.x * val.x + val.y * val.y + val.z * val.z + val.w * val.w;
#pragma unroll
  for (int m = 16; m >= 1; m >>= 1) {
    s  += __shfl_xor(s, m);
    s2 += __shfl_xor(s2, m);
  }
  __shared__ float sh[2][8];
  int wave = tid >> 5, lane = tid & 31;
  if (lane == 0) { sh[0][wave] = s; sh[1][wave] = s2; }
  __syncthreads();
  if (tid == 0) {
    float a = 0.f, b2 = 0.f;
#pragma unroll
    for (int i = 0; i < 8; ++i) { a += sh[0][i]; b2 += sh[1][i]; }
    sh[0][0] = a; sh[1][0] = b2;
  }
  __syncthreads();
  float mean = sh[0][0] * (1.0f / C);
  float var  = sh[1][0] * (1.0f / C) - mean * mean;
  float rstd = rsqrtf(var + 1e-5f);
  int c0 = tid * 4;
  float vv[4] = {val.x, val.y, val.z, val.w};
#pragma unroll
  for (int j = 0; j < 4; ++j) {
    out[(size_t)row * C + c0 + j] =
        f2bf((vv[j] - mean) * rstd * g[c0 + j] + be[c0 + j]);
  }
}

// ---------------------------------------------------------------------------
// WMMA bf16 GEMM: out[m,n] = sum_k A[m,k] * W[n,k]   (W stored [N,K])
// Block tile 128x128, K-step 32. 8 waves: wave(wm 0..3 -> 32 rows,
// wn 0..1 -> 64 cols); each wave: 2x4 = 8 accumulators, 8 WMMAs / K-step.
// Tiles staged into LDS by the Tensor Data Mover (one D# per tile) when
// available, else by cooperative b128 vector loads.
// MODE 0: outB = bf16(acc)
// MODE 1: outF = resid + acc
// MODE 2: outB = bf16(gelu(acc + bias[n]))     (exact erf GELU)
// MODE 3: outF = resid + acc + bias[n]
// ---------------------------------------------------------------------------
template <int MODE>
__global__ __launch_bounds__(256) void gemm_bf16_kernel(
    const __bf16* __restrict__ A, const __bf16* __restrict__ W,
    const float* __restrict__ bias, const float* __restrict__ resid,
    float* __restrict__ outF, __bf16* __restrict__ outB,
    int M, int N, int K) {
  __shared__ __bf16 As[128][32];   // 8 KB
  __shared__ __bf16 Ws[128][32];   // 8 KB
  int tid  = threadIdx.x;
  int lane = tid & 31, wave = tid >> 5;
  int wm = wave & 3, wn = wave >> 2;
  int l15 = lane & 15, lh = lane >> 4;
  int mBase = blockIdx.y * 128, nBase = blockIdx.x * 128;

  v8f acc[2][4] = {{{}, {}, {}, {}}, {{}, {}, {}, {}}};
  int kb8 = lh * 8;
  int kh  = lh * 16;
#if !USE_TDM
  int lr = tid >> 1;           // 0..127
  int lc = (tid & 1) * 16;     // 0 or 16
#endif
#if USE_TDM
  unsigned ldsA = (unsigned)(size_t)(void*)&As[0][0];
  unsigned ldsW = (unsigned)(size_t)(void*)&Ws[0][0];
#endif

  for (int k0 = 0; k0 < K; k0 += 32) {
#if USE_TDM
    if (wave == 0) {
      // TDM: DMA 128x32 bf16 tiles into LDS; tracked on TENSORcnt.
      tdm_load_2d(ldsA, A + (size_t)mBase * K + k0, (unsigned)K, (unsigned)M,
                  32u, 128u, (unsigned)K);
      tdm_load_2d(ldsW, W + (size_t)nBase * K + k0, (unsigned)K, (unsigned)N,
                  32u, 128u, (unsigned)K);
      __builtin_amdgcn_s_wait_tensorcnt(0);
    }
#else
    *(uint4*)&As[lr][lc] =
        *(const uint4*)&A[(size_t)(mBase + lr) * K + k0 + lc];
    *(uint4*)&As[lr][lc + 8] =
        *(const uint4*)&A[(size_t)(mBase + lr) * K + k0 + lc + 8];
    *(uint4*)&Ws[lr][lc] =
        *(const uint4*)&W[(size_t)(nBase + lr) * K + k0 + lc];
    *(uint4*)&Ws[lr][lc + 8] =
        *(const uint4*)&W[(size_t)(nBase + lr) * K + k0 + lc + 8];
#endif
    __syncthreads();

    Frag a[2], b[4];
#pragma unroll
    for (int mi = 0; mi < 2; ++mi) {
      int am = wm * 32 + mi * 16 + l15;
      a[mi].q[0] = *(const uint4*)&As[am][kb8];
      a[mi].q[1] = *(const uint4*)&As[am][kb8 + 16];
    }
#pragma unroll
    for (int ni = 0; ni < 4; ++ni) {
      int bn = wn * 64 + ni * 16 + l15;
      b[ni].q[0] = *(const uint4*)&Ws[bn][kb8];
      b[ni].q[1] = *(const uint4*)&Ws[bn][kb8 + 16];
    }
#pragma unroll
    for (int mi = 0; mi < 2; ++mi)
#pragma unroll
      for (int ni = 0; ni < 4; ++ni)
        acc[mi][ni] = wmma_bf16(a[mi], b[ni], acc[mi][ni]);
    __syncthreads();
  }

#pragma unroll
  for (int mi = 0; mi < 2; ++mi) {
    int mRow = mBase + wm * 32 + mi * 16 + 8 * lh;
#pragma unroll
    for (int ni = 0; ni < 4; ++ni) {
      int n = nBase + wn * 64 + ni * 16 + l15;
#pragma unroll
      for (int v = 0; v < 8; ++v) {
        int m = mRow + v;
        float xv = acc[mi][ni][v];
        size_t i0 = (size_t)m * N + n;
        if (MODE == 0) {
          outB[i0] = f2bf(xv);
        } else if (MODE == 1) {
          outF[i0] = resid[i0] + xv;
        } else if (MODE == 2) {
          float gv = xv + bias[n];
          gv = 0.5f * gv * (1.0f + erff(gv * 0.70710678118654752f));
          outB[i0] = f2bf(gv);
        } else {  // MODE 3
          outF[i0] = resid[i0] + xv + bias[n];
        }
      }
    }
  }
}

// ---------------------------------------------------------------------------
// Flash-style causal attention with WMMA.
// Q,K: bf16 [B*T, H*64] (row = token). Vt: bf16 [B*H*64, T].
// O:   bf16 [B*T, H*64].
// grid = (T/128, H, B); 8 waves/block; each wave owns one 16-query tile and
// streams 32 keys/step with online softmax.
// ---------------------------------------------------------------------------
__global__ __launch_bounds__(256) void attn_kernel(
    const __bf16* __restrict__ Q, const __bf16* __restrict__ Kb,
    const __bf16* __restrict__ Vt, __bf16* __restrict__ O) {
  __shared__ __bf16 Plds[8][16 * 32];   // per-wave P staging (C-layout -> A-layout)
  const int C = CEMB;
  int lane = threadIdx.x & 31, wave = threadIdx.x >> 5;
  int l15 = lane & 15, lh = lane >> 4;
  int b = blockIdx.z, h = blockIdx.y;
  int qBase = (blockIdx.x * 8 + wave) * 16;
  int qEnd  = qBase + 15;
  const float scale = 0.125f;            // 1/sqrt(64)

  // Q fragments (d 0..31 and 32..63), loaded once.
  size_t qrow = ((size_t)(b * TSEQ + qBase + l15)) * C + h * HDIM;
  int kb8 = lh * 8;
  Frag aq0, aq1;
  aq0.q[0] = *(const uint4*)&Q[qrow + kb8];
  aq0.q[1] = *(const uint4*)&Q[qrow + kb8 + 16];
  aq1.q[0] = *(const uint4*)&Q[qrow + 32 + kb8];
  aq1.q[1] = *(const uint4*)&Q[qrow + 32 + kb8 + 16];

  v8f o[4] = {{}, {}, {}, {}};
  float mst[8], lst[8];
#pragma unroll
  for (int v = 0; v < 8; ++v) { mst[v] = -1e30f; lst[v] = 0.f; }

  __bf16* pw = &Plds[wave][0];
  size_t vhead = (size_t)(b * NHEAD + h) * HDIM;   // row base into Vt

  for (int kt = 0; kt <= qEnd; kt += 32) {
    // ---- S = Q K^T for keys [kt, kt+32) : two 16x16 logit tiles ----
    v8f s0 = {}, s1 = {};
    size_t kr0 = ((size_t)(b * TSEQ + kt + l15)) * C + h * HDIM + lh * 8;
    size_t kr1 = kr0 + (size_t)16 * C;
    Frag bk;
    bk.q[0] = *(const uint4*)&Kb[kr0];
    bk.q[1] = *(const uint4*)&Kb[kr0 + 16];
    s0 = wmma_bf16(aq0, bk, s0);
    bk.q[0] = *(const uint4*)&Kb[kr0 + 32];
    bk.q[1] = *(const uint4*)&Kb[kr0 + 48];
    s0 = wmma_bf16(aq1, bk, s0);
    bk.q[0] = *(const uint4*)&Kb[kr1];
    bk.q[1] = *(const uint4*)&Kb[kr1 + 16];
    s1 = wmma_bf16(aq0, bk, s1);
    bk.q[0] = *(const uint4*)&Kb[kr1 + 32];
    bk.q[1] = *(const uint4*)&Kb[kr1 + 48];
    s1 = wmma_bf16(aq1, bk, s1);

    // ---- scale + causal mask + online softmax update ----
    float alpha[8];
#pragma unroll
    for (int v = 0; v < 8; ++v) {
      int mrow = qBase + v + 8 * lh;
      int key0 = kt + l15;
      float t0 = (key0      <= mrow) ? s0[v] * scale : -1e30f;
      float t1 = (key0 + 16 <= mrow) ? s1[v] * scale : -1e30f;
      float rm = fmaxf(t0, t1);
#pragma unroll
      for (int msk = 8; msk >= 1; msk >>= 1) rm = fmaxf(rm, __shfl_xor(rm, msk));
      float mnew = fmaxf(mst[v], rm);
      float a = __expf(mst[v] - mnew);
      float p0 = __expf(t0 - mnew);
      float p1 = __expf(t1 - mnew);
      float rs = p0 + p1;
#pragma unroll
      for (int msk = 8; msk >= 1; msk >>= 1) rs += __shfl_xor(rs, msk);
      mst[v] = mnew;
      lst[v] = lst[v] * a + rs;
      alpha[v] = a;
      s0[v] = p0;
      s1[v] = p1;
    }
    // rescale existing accumulator
#pragma unroll
    for (int nc = 0; nc < 4; ++nc)
#pragma unroll
      for (int v = 0; v < 8; ++v) o[nc][v] *= alpha[v];

    // ---- P: C-layout -> LDS -> A-fragment layout ----
#pragma unroll
    for (int v = 0; v < 8; ++v) {
      int r = v + 8 * lh;
      pw[r * 32 + l15]      = f2bf(s0[v]);
      pw[r * 32 + 16 + l15] = f2bf(s1[v]);
    }
    Frag ap;  // same-wave LDS ops are in-order; compiler inserts dscnt waits
    ap.q[0] = *(const uint4*)&pw[l15 * 32 + kb8];
    ap.q[1] = *(const uint4*)&pw[l15 * 32 + kb8 + 16];

    // ---- O += P V : 4 WMMAs over d-chunks, Vt rows contiguous in t ----
#pragma unroll
    for (int nc = 0; nc < 4; ++nc) {
      int d = nc * 16 + l15;
      size_t va = (vhead + d) * TSEQ + kt + lh * 8;
      Frag bv;
      bv.q[0] = *(const uint4*)&Vt[va];
      bv.q[1] = *(const uint4*)&Vt[va + 16];
      o[nc] = wmma_bf16(ap, bv, o[nc]);
    }
  }

  // ---- finalize and store [B,T,H,D] (== row-major [M, C]) ----
#pragma unroll
  for (int v = 0; v < 8; ++v) {
    float inv = 1.0f / lst[v];
    size_t row = (size_t)(b * TSEQ + qBase + v + 8 * lh);
#pragma unroll
    for (int nc = 0; nc < 4; ++nc) {
      O[row * C + h * HDIM + nc * 16 + l15] = f2bf(o[nc][v] * inv);
    }
  }
}

// ---------------------------------------------------------------------------
// Host-side orchestration
// ---------------------------------------------------------------------------
extern "C" void kernel_launch(void* const* d_in, const int* in_sizes, int n_in,
                              void* d_out, int out_size, void* d_ws, size_t ws_size,
                              hipStream_t stream) {
  const float* x   = (const float*)d_in[0];
  const float* Wq  = (const float*)d_in[1];
  const float* Wk  = (const float*)d_in[2];
  const float* Wv  = (const float*)d_in[3];
  const float* Wo  = (const float*)d_in[4];
  const float* W1  = (const float*)d_in[5];
  const float* b1  = (const float*)d_in[6];
  const float* W2  = (const float*)d_in[7];
  const float* b2  = (const float*)d_in[8];
  const float* g1  = (const float*)d_in[9];
  const float* be1 = (const float*)d_in[10];
  const float* g2  = (const float*)d_in[11];
  const float* be2 = (const float*)d_in[12];

  char* w = (char*)d_ws;
  const size_t MB = 1ull << 20;
  __bf16* Wq_b  = (__bf16*)(w + 0 * MB);     // 2 MB each
  __bf16* Wk_b  = (__bf16*)(w + 2 * MB);
  __bf16* Wv_b  = (__bf16*)(w + 4 * MB);
  __bf16* Wo_b  = (__bf16*)(w + 6 * MB);
  __bf16* W1t   = (__bf16*)(w + 8 * MB);     // 8 MB   [4096,1024]
  __bf16* W2t   = (__bf16*)(w + 16 * MB);    // 8 MB   [1024,4096]
  __bf16* hb    = (__bf16*)(w + 24 * MB);    // 16 MB  LN1 out
  __bf16* qb    = (__bf16*)(w + 40 * MB);    // 16 MB
  __bf16* kbuf  = (__bf16*)(w + 56 * MB);    // 16 MB
  __bf16* vbuf  = (__bf16*)(w + 72 * MB);    // 16 MB
  __bf16* vt    = (__bf16*)(w + 88 * MB);    // 16 MB
  float*  x1    = (float*)(w + 104 * MB);    // 32 MB  attn residual sum
  __bf16* attnb = hb;                         // reuse (hb dead after QKV GEMMs)
  __bf16* h2b   = (__bf16*)(w + 24 * MB);     // reuse (attnb dead after Wo GEMM)
  __bf16* ffb   = (__bf16*)(w + 40 * MB);     // 64 MB reuse q/k/v/vt (dead)

  const int C = CEMB, M = MTOT;
  const int CC = C * C;           // 1M elems
  dim3 blk(256);

  // weight conversion
  f32_to_bf16_kernel<<<(CC + 255) / 256, blk, 0, stream>>>(Wq, Wq_b, CC);
  f32_to_bf16_kernel<<<(CC + 255) / 256, blk, 0, stream>>>(Wk, Wk_b, CC);
  f32_to_bf16_kernel<<<(CC + 255) / 256, blk, 0, stream>>>(Wv, Wv_b, CC);
  f32_to_bf16_kernel<<<(CC + 255) / 256, blk, 0, stream>>>(Wo, Wo_b, CC);
  f32_to_bf16_T_kernel<<<(4 * CC + 255) / 256, blk, 0, stream>>>(W1, W1t, C, 4 * C);
  f32_to_bf16_T_kernel<<<(4 * CC + 255) / 256, blk, 0, stream>>>(W2, W2t, 4 * C, C);

  // LN1
  ln_bf16_kernel<<<M, blk, 0, stream>>>(x, g1, be1, hb);

  // QKV projections (h @ W.T)
  dim3 gproj(C / 128, M / 128);
  gemm_bf16_kernel<0><<<gproj, blk, 0, stream>>>(hb, Wq_b, nullptr, nullptr,
                                                 nullptr, qb, M, C, C);
  gemm_bf16_kernel<0><<<gproj, blk, 0, stream>>>(hb, Wk_b, nullptr, nullptr,
                                                 nullptr, kbuf, M, C, C);
  gemm_bf16_kernel<0><<<gproj, blk, 0, stream>>>(hb, Wv_b, nullptr, nullptr,
                                                 nullptr, vbuf, M, C, C);

  // V transpose for contiguous B-fragments in PV
  transpose_v_kernel<<<(M * C + 255) / 256, blk, 0, stream>>>(vbuf, vt);

  // causal attention
  dim3 gattn(TSEQ / 128, NHEAD, BATCH);
  attn_kernel<<<gattn, blk, 0, stream>>>(qb, kbuf, vt, attnb);

  // x1 = x + attn @ Wo.T
  gemm_bf16_kernel<1><<<gproj, blk, 0, stream>>>(attnb, Wo_b, nullptr, x,
                                                 x1, nullptr, M, C, C);

  // LN2
  ln_bf16_kernel<<<M, blk, 0, stream>>>(x1, g2, be2, h2b);

  // FFN
  dim3 gff1(4 * C / 128, M / 128);
  gemm_bf16_kernel<2><<<gff1, blk, 0, stream>>>(h2b, W1t, b1, nullptr,
                                                nullptr, ffb, M, 4 * C, C);
  gemm_bf16_kernel<3><<<gproj, blk, 0, stream>>>(ffb, W2t, b2, x1,
                                                 (float*)d_out, nullptr,
                                                 M, C, 4 * C);
}
